// GNN_8830452760606
// MI455X (gfx1250) — matrix operand order repaired and checked by reading, not compiled, not run on hardware
//
#include <hip/hip_runtime.h>
#include <hip/hip_bf16.h>

// ---------------------------------------------------------------------------
// GCN on MI455X (gfx1250): edge scatter converted to dense normalized
// adjacency matmul so every heavy op runs on bf16 WMMA (16x16x32, f32 acc).
// B operand staged TRANSPOSED in LDS so fragments are contiguous b128 reads.
// ---------------------------------------------------------------------------

typedef __attribute__((ext_vector_type(16))) __bf16 v16bf;
typedef __attribute__((ext_vector_type(8)))  __bf16 v8bf;
typedef __attribute__((ext_vector_type(8)))  float  v8f;

#define GNN_N  2048
#define GNN_IN 64
#define GNN_H  4096
#define GNN_E  65536

// ------------------------------- helpers -----------------------------------

__global__ void k_zero_f32(float* __restrict__ p, int n) {
    int i = blockIdx.x * blockDim.x + threadIdx.x;
    int stride = gridDim.x * blockDim.x;
    for (; i < n; i += stride) p[i] = 0.0f;
}

__global__ void k_degree(const int* __restrict__ src, const int* __restrict__ dst,
                         float* __restrict__ deg_out, float* __restrict__ deg_in, int e) {
    int i = blockIdx.x * blockDim.x + threadIdx.x;
    if (i < e) {
        atomicAdd(&deg_out[src[i]], 1.0f);
        atomicAdd(&deg_in[dst[i]], 1.0f);
    }
}

__global__ void k_norm(const float* __restrict__ deg, float* __restrict__ nrm, int n) {
    int i = blockIdx.x * blockDim.x + threadIdx.x;
    if (i < n) nrm[i] = rsqrtf(fmaxf(deg[i], 1.0f));
}

// Dense normalized adjacency: adj[d*n+s] += norm_dst[d] * norm_src[s] per edge.
// Multi-edges accumulate, exactly matching segment_sum semantics.
__global__ void k_build_adj(const int* __restrict__ src, const int* __restrict__ dst,
                            const float* __restrict__ ns, const float* __restrict__ nd,
                            float* __restrict__ adj, int e, int n) {
    int i = blockIdx.x * blockDim.x + threadIdx.x;
    if (i < e) {
        int s = src[i], d = dst[i];
        atomicAdd(&adj[(size_t)d * n + s], nd[d] * ns[s]);
    }
}

__global__ void k_f32_to_bf16(const float* __restrict__ x, __bf16* __restrict__ y, int n) {
    int i = blockIdx.x * blockDim.x + threadIdx.x;
    int stride = gridDim.x * blockDim.x;
    for (; i < n; i += stride) y[i] = (__bf16)x[i];
}

// --------------------------- WMMA bf16 GEMM --------------------------------
// C[M,N] = op(A[M,K] @ B[K,N] (+bias) (relu)) -> bf16.
// Block = 256 threads = 8 waves arranged WMW x WNW; each wave owns 1 M-tile
// and TN N-tiles (A fragment reused). K staged 64-deep per LDS fill, with B
// stored transposed (BT[col][k], padded rows) so every lane's fragment is a
// contiguous, 16B-aligned 32-byte LDS read -> 2x ds_load_b128, no packing.
template<int WMW, int WNW, int TN, bool BIAS, bool RELU>
__global__ __launch_bounds__(256)
void k_gemm_bf16(const __bf16* __restrict__ A, const __bf16* __restrict__ B,
                 const float* __restrict__ bias, __bf16* __restrict__ C,
                 int M, int N, int K) {
    constexpr int BN  = WNW * TN * 16;   // block columns
    constexpr int BK  = 64;              // k-depth per stage
    constexpr int LDT = BK + 8;          // 72 elems = 144 B row (16B multiple)
    __shared__ __bf16 ldsBT[BN * LDT];   // BT[col][k]

    const int lane  = threadIdx.x & 31;
    const int wave  = threadIdx.x >> 5;
    const int waveN = wave % WNW;
    const int waveM = wave / WNW;
    const int r     = lane & 15;
    const int half  = lane >> 4;

    const int colBlk = blockIdx.x * BN;
    const int row0   = (blockIdx.y * WMW + waveM) * 16;
    const int col0w  = colBlk + waveN * TN * 16;

    v8f acc[TN];
#pragma unroll
    for (int t = 0; t < TN; t++) acc[t] = (v8f){};

    for (int k0 = 0; k0 < K; k0 += BK) {
        // stage B[k0..k0+63][colBlk..+BN-1], transposed, coalesced 8B reads
        for (int q = threadIdx.x; q < (BK * BN) / 4; q += 256) {
            int e  = q * 4;
            int kr = e / BN, cc = e % BN;
            union { uint2 u; __bf16 h[4]; } cv;
            cv.u = *(const uint2*)&B[(size_t)(k0 + kr) * N + colBlk + cc];
#pragma unroll
            for (int j = 0; j < 4; j++)
                ldsBT[(cc + j) * LDT + kr] = cv.h[j];
        }
        __syncthreads();

#pragma unroll
        for (int ks = 0; ks < 2; ks++) {
            const int kk = k0 + ks * 32;
            // A fragment: row (row0+r); lanes 0-15 hold K {0..15},
            // lanes 16-31 hold K {16..31} of this 32-slice.
            const __bf16* ap = A + (size_t)(row0 + r) * K + kk + half * 16;
            v8bf alo = *(const v8bf*)ap;          // global_load_b128
            v8bf ahi = *(const v8bf*)(ap + 8);    // global_load_b128
            v16bf a;
#pragma unroll
            for (int i = 0; i < 8; i++) { a[i] = alo[i]; a[8 + i] = ahi[i]; }
            __builtin_prefetch((const void*)(ap + BK), 0, 1);  // next K stage

#pragma unroll
            for (int t = 0; t < TN; t++) {
                // column (col0w + t*16 + r), k rows kk+half*16 .. +15:
                // contiguous 32B in LDS, 16B aligned -> 2x ds_load_b128
                const __bf16* bp =
                    ldsBT + (waveN * TN * 16 + t * 16 + r) * LDT + ks * 32 + half * 16;
                v16bf b = *(const v16bf*)bp;
                acc[t] = __builtin_amdgcn_wmma_f32_16x16x32_bf16(
                    /*neg_a=*/false, a, /*neg_b=*/false, b,
                    /*c_mod=*/(short)0, acc[t],
                    /*reuse_a=*/false, /*reuse_b=*/false);
            }
        }
        __syncthreads();
    }

    // C/D layout: lanes 0-15 -> rows 0..7, lanes 16-31 -> rows 8..15; col = r
#pragma unroll
    for (int t = 0; t < TN; t++) {
        const int col = col0w + t * 16 + r;
        float bv = BIAS ? bias[col] : 0.0f;
#pragma unroll
        for (int j = 0; j < 8; j++) {
            float v = acc[t][j] + bv;
            if (RELU) v = fmaxf(v, 0.0f);
            C[(size_t)(row0 + half * 8 + j) * N + col] = (__bf16)v;
        }
    }
}

// Final head: out[i] = M2[i,:] . Wl3 + bl3  (K=16, N=1 — pure VALU, trivial)
__global__ void k_head(const __bf16* __restrict__ M2, const float* __restrict__ Wl3,
                       const float* __restrict__ bl3, float* __restrict__ out, int n) {
    int i = blockIdx.x * blockDim.x + threadIdx.x;
    if (i < n) {
        float s = bl3[0];
#pragma unroll
        for (int k = 0; k < 16; k++) s += (float)M2[i * 16 + k] * Wl3[k];
        out[i] = s;
    }
}

// ------------------------------- launch ------------------------------------

extern "C" void kernel_launch(void* const* d_in, const int* in_sizes, int n_in,
                              void* d_out, int out_size, void* d_ws, size_t ws_size,
                              hipStream_t stream) {
    (void)in_sizes; (void)n_in; (void)out_size; (void)ws_size;

    const float* features = (const float*)d_in[0];
    const float* weight   = (const float*)d_in[1];
    const float* W1       = (const float*)d_in[2];
    const float* b1       = (const float*)d_in[3];
    const float* Wl1      = (const float*)d_in[4];
    const float* bl1      = (const float*)d_in[5];
    const float* Wl2      = (const float*)d_in[6];
    const float* bl2      = (const float*)d_in[7];
    const float* Wl3      = (const float*)d_in[8];
    const float* bl3      = (const float*)d_in[9];
    const int*   src      = (const int*)d_in[10];
    const int*   dst      = (const int*)d_in[11];
    float* out = (float*)d_out;

    char* ws = (char*)d_ws;
    size_t off = 0;
    auto alloc = [&](size_t bytes) -> void* {
        void* p = ws + off;
        off = (off + bytes + 255) & ~(size_t)255;
        return p;
    };

    float*  deg_out = (float*)alloc((size_t)GNN_N * 4);
    float*  deg_in  = (float*)alloc((size_t)GNN_N * 4);
    float*  ns      = (float*)alloc((size_t)GNN_N * 4);
    float*  nd      = (float*)alloc((size_t)GNN_N * 4);
    float*  adjf    = (float*)alloc((size_t)GNN_N * GNN_N * 4);   // 16 MB
    __bf16* adjb    = (__bf16*)alloc((size_t)GNN_N * GNN_N * 2);  // 8 MB
    __bf16* Fb      = (__bf16*)alloc((size_t)GNN_N * GNN_IN * 2);
    __bf16* W1b     = (__bf16*)alloc((size_t)GNN_IN * GNN_H * 2);
    __bf16* Wb      = (__bf16*)alloc((size_t)GNN_H * GNN_N * 2);  // 16 MB
    __bf16* Wl1b    = (__bf16*)alloc((size_t)GNN_N * 64 * 2);
    __bf16* Wl2b    = (__bf16*)alloc((size_t)64 * 16 * 2);
    __bf16* Tb      = (__bf16*)alloc((size_t)GNN_N * GNN_IN * 2);
    __bf16* Xb      = (__bf16*)alloc((size_t)GNN_N * GNN_H * 2);  // 16 MB
    __bf16* Yb      = (__bf16*)alloc((size_t)GNN_N * GNN_N * 2);  // 8 MB
    __bf16* Zb      = (__bf16*)alloc((size_t)GNN_N * GNN_N * 2);  // 8 MB
    __bf16* M1b     = (__bf16*)alloc((size_t)GNN_N * 64 * 2);
    __bf16* M2b     = (__bf16*)alloc((size_t)GNN_N * 16 * 2);

    // ---- degrees, norms, dense normalized adjacency ----
    k_zero_f32<<<32, 256, 0, stream>>>(deg_out, GNN_N);
    k_zero_f32<<<32, 256, 0, stream>>>(deg_in,  GNN_N);
    k_zero_f32<<<2048, 256, 0, stream>>>(adjf, GNN_N * GNN_N);
    k_degree<<<GNN_E / 256, 256, 0, stream>>>(src, dst, deg_out, deg_in, GNN_E);
    k_norm<<<8, 256, 0, stream>>>(deg_out, ns, GNN_N);
    k_norm<<<8, 256, 0, stream>>>(deg_in,  nd, GNN_N);
    k_build_adj<<<GNN_E / 256, 256, 0, stream>>>(src, dst, ns, nd, adjf, GNN_E, GNN_N);

    // ---- bf16 operand conversions ----
    k_f32_to_bf16<<<2048, 256, 0, stream>>>(adjf, adjb, GNN_N * GNN_N);
    k_f32_to_bf16<<<512,  256, 0, stream>>>(features, Fb, GNN_N * GNN_IN);
    k_f32_to_bf16<<<1024, 256, 0, stream>>>(W1, W1b, GNN_IN * GNN_H);
    k_f32_to_bf16<<<4096, 256, 0, stream>>>(weight, Wb, GNN_H * GNN_N);
    k_f32_to_bf16<<<512,  256, 0, stream>>>(Wl1, Wl1b, GNN_N * 64);
    k_f32_to_bf16<<<4,    256, 0, stream>>>(Wl2, Wl2b, 64 * 16);

    // ---- WMMA GEMM chain (block tile 64x64: 4x2 waves, TN=2) ----
    // T = Adj @ F                  [2048 x 64],  K=2048
    k_gemm_bf16<4, 2, 2, false, false>
        <<<dim3(1, GNN_N / 64), 256, 0, stream>>>(adjb, Fb, nullptr, Tb,
                                                  GNN_N, GNN_IN, GNN_N);
    // X = relu(T @ W1 + b1)        [2048 x 4096], K=64
    k_gemm_bf16<4, 2, 2, true, true>
        <<<dim3(GNN_H / 64, GNN_N / 64), 256, 0, stream>>>(Tb, W1b, b1, Xb,
                                                           GNN_N, GNN_H, GNN_IN);
    // Y = X @ weight               [2048 x 2048], K=4096  (dominant, 34 GFLOP)
    k_gemm_bf16<4, 2, 2, false, false>
        <<<dim3(GNN_N / 64, GNN_N / 64), 256, 0, stream>>>(Xb, Wb, nullptr, Yb,
                                                           GNN_N, GNN_N, GNN_H);
    // Z = relu(Adj @ Y)            [2048 x 2048], K=2048  (edge aggregation)
    k_gemm_bf16<4, 2, 2, false, true>
        <<<dim3(GNN_N / 64, GNN_N / 64), 256, 0, stream>>>(adjb, Yb, nullptr, Zb,
                                                           GNN_N, GNN_N, GNN_N);
    // M1 = relu(Z @ Wl1 + bl1)     [2048 x 64], K=2048
    k_gemm_bf16<4, 2, 2, true, true>
        <<<dim3(1, GNN_N / 64), 256, 0, stream>>>(Zb, Wl1b, bl1, M1b,
                                                  GNN_N, 64, GNN_N);
    // M2 = relu(M1 @ Wl2 + bl2)    [2048 x 16], K=64
    k_gemm_bf16<8, 1, 1, true, true>
        <<<dim3(1, GNN_N / 128), 256, 0, stream>>>(M1b, Wl2b, bl2, M2b,
                                                   GNN_N, 16, 64);
    // out = M2 @ Wl3 + bl3         [2048 x 1]
    k_head<<<GNN_N / 256, 256, 0, stream>>>(M2b, Wl3, bl3, out, GNN_N);
}
